// PDAttention_19731079757906
// MI455X (gfx1250) — matrix operand order repaired and checked by reading, not compiled, not run on hardware
//
#include <hip/hip_runtime.h>
#include <hip/hip_bf16.h>
#include <stdint.h>

typedef __attribute__((ext_vector_type(16))) __bf16 v16bf;
typedef __attribute__((ext_vector_type(8)))  float  v8f;
typedef __attribute__((ext_vector_type(4)))  float  f4;   // native vector for NT loads

union ABPack { uint32_t u[8]; v16bf v; };

// Two f32 -> packed bf16 pair by truncation: one v_perm_b32.
__device__ __forceinline__ uint32_t pk_bf16_trunc(float lo, float hi) {
  return __builtin_amdgcn_perm(__float_as_uint(hi), __float_as_uint(lo), 0x07060302u);
}

// ---------------------------------------------------------------------------
// Stage 1: per batch, Q[b] = ctxt[b]·Wq^T + bq;  s[b] = Q[b]·bk;
//          q~[b,m] = sum_k Q[b,k] * Wk[k,m], stored as bf16 (RNE).
// Tiny kernel (B blocks x 128 threads), negligible runtime.
// ---------------------------------------------------------------------------
__global__ void pda_qproj(const float* __restrict__ ctxt,
                          const float* __restrict__ Wq,
                          const float* __restrict__ bq,
                          const float* __restrict__ Wk,
                          const float* __restrict__ bk,
                          uint16_t* __restrict__ qkt,  // [B, dm] bf16
                          float* __restrict__ sb,      // [B]
                          int dc, int dk, int dm) {
  const int b = blockIdx.x;
  const int t = threadIdx.x;            // 0..dk-1 (dk == dm == 128)
  extern __shared__ float sm1[];
  float* Qs  = sm1;                     // dk
  float* red = sm1 + dk;                // dk

  const float* cb = ctxt + (size_t)b * dc;
  const float* wr = Wq + (size_t)t * dc;
  float acc = bq[t];
  for (int c = 0; c < dc; ++c) acc += cb[c] * wr[c];
  Qs[t]  = acc;
  red[t] = acc * bk[t];
  __syncthreads();

  if (t == 0) {
    float s = 0.f;
    for (int k = 0; k < dk; ++k) s += red[k];
    sb[b] = s;
  }

  float qt = 0.f;
  for (int k = 0; k < dk; ++k) qt += Qs[k] * Wk[(size_t)k * dm + t];
  uint32_t ub = __float_as_uint(qt);
  ub += 0x7FFFu + ((ub >> 16) & 1u);    // RNE here (cheap, tiny kernel)
  qkt[(size_t)b * dm + t] = (uint16_t)(ub >> 16);
}

// ---------------------------------------------------------------------------
// Stage 2: one block per batch. Pass 1 streams emb[b] in 16-row tiles through
// v_wmma_f32_16x16x32_bf16 (q~ in B column 0) and stores RAW dots to LDS.
// Passes 2-4 are full-width coalesced: tanh clip + mask + max, exp + sum,
// scaled write. Bandwidth-bound: 2.56 MB of emb per block, NT-hinted.
// ---------------------------------------------------------------------------
__global__ void __launch_bounds__(512)
pda_scores_softmax(const float* __restrict__ emb,
                   const float* __restrict__ mask,
                   const uint16_t* __restrict__ qkt,
                   const float* __restrict__ sb,
                   float* __restrict__ out,
                   int N, int dm /*must be 128*/) {
  const int b      = blockIdx.x;
  const int tid    = threadIdx.x;
  const int lane   = tid & 31;
  const int wid    = tid >> 5;
  const int nwaves = blockDim.x >> 5;
  const int numTiles = (N + 15) >> 4;
  const int padN     = numTiles << 4;

  extern __shared__ float sm2[];
  float*    u_s  = sm2;                              // padN scores
  uint32_t* qk_s = (uint32_t*)(sm2 + padN);          // dm/2 bf16-pairs
  float*    red  = sm2 + padN + (dm >> 1);           // nwaves partials

  // Stage q~ (bf16) into LDS once.
  for (int i = tid; i < (dm >> 1); i += blockDim.x)
    qk_s[i] = ((const uint32_t*)qkt)[(size_t)b * (dm >> 1) + i];
  __syncthreads();

  // B operand: 32x16 bf16, column 0 = q~ chunk, other columns zero.
  // Layout: lane c (c<16) holds K=2j,2j+1 in VGPR j; lane c+16 holds K=16+2j.
  const int  half  = lane >> 4;          // 0 -> K 0..15 of chunk, 1 -> K 16..31
  const bool bcol0 = (lane & 15) == 0;   // lanes owning column 0
  ABPack bm[4];
#pragma unroll
  for (int j = 0; j < 4; ++j)
#pragma unroll
    for (int v = 0; v < 8; ++v)
      bm[j].u[v] = bcol0 ? qk_s[j * 16 + half * 8 + v] : 0u;

  const int row0 = lane & 15;
  const int koff = half * 16;            // float offset inside 32-col chunk

  // ---- Pass 1: WMMA dots -> raw scores in LDS -----------------------------
  for (int t = wid; t < numTiles; t += nwaves) {
    const int nbase = t << 4;
    int row = nbase + row0;
    if (row >= N) row = N - 1;           // clamp pad rows (fixed up in pass 2)
    const float* rp = emb + ((size_t)b * N + row) * dm;

    v8f c = {};
#pragma unroll
    for (int j = 0; j < 4; ++j) {
      // A-matrix 16x32 bf16: lane<16 holds K=koff..koff+15 (V0..7)
      // of this 32-wide chunk; lane>=16 koff=16.
      const f4* p = (const f4*)(rp + j * 32 + koff);
      f4 f0 = __builtin_nontemporal_load(p);
      f4 f1 = __builtin_nontemporal_load(p + 1);
      f4 f2 = __builtin_nontemporal_load(p + 2);   // +8 floats
      f4 f3 = __builtin_nontemporal_load(p + 3);
      ABPack a;
      a.u[0] = pk_bf16_trunc(f0.x, f0.y);
      a.u[1] = pk_bf16_trunc(f0.z, f0.w);
      a.u[2] = pk_bf16_trunc(f1.x, f1.y);
      a.u[3] = pk_bf16_trunc(f1.z, f1.w);
      a.u[4] = pk_bf16_trunc(f2.x, f2.y);
      a.u[5] = pk_bf16_trunc(f2.z, f2.w);
      a.u[6] = pk_bf16_trunc(f3.x, f3.y);
      a.u[7] = pk_bf16_trunc(f3.z, f3.w);
      c = __builtin_amdgcn_wmma_f32_16x16x32_bf16(false, a.v, false, bm[j].v,
                                                  (short)0, c, false, false);
    }

    // Column 0 of D: lane 0 -> rows 0..7, lane 16 -> rows 8..15.
    // Store raw dots only (all transcendental work deferred to pass 2).
    if (bcol0) {
      const int n0 = nbase + half * 8;
      f4 lo = {c[0], c[1], c[2], c[3]};
      f4 hi = {c[4], c[5], c[6], c[7]};
      *(f4*)(u_s + n0)     = lo;        // ds_store_b128
      *(f4*)(u_s + n0 + 4) = hi;
    }
  }
  __syncthreads();

  // ---- Pass 2: coalesced tanh clip + mask + running max -------------------
  const float sbias = sb[b];
  const float scale = 0.08838834764831845f;   // 1/sqrt(128)
  float lmax = -3.0e38f;
  for (int n = tid; n < padN; n += blockDim.x) {
    float u;
    if (n < N) {
      float mk = __builtin_nontemporal_load(&mask[(size_t)b * N + n]);
      u = 10.0f * tanhf((u_s[n] + sbias) * scale) + mk;
    } else {
      u = -3.0e38f;                      // pad slots excluded from softmax
    }
    u_s[n] = u;
    lmax = fmaxf(lmax, u);
  }
  __syncthreads();

  // Block max reduction.
  float m = lmax;
#pragma unroll
  for (int off = 16; off > 0; off >>= 1)
    m = fmaxf(m, __shfl_xor(m, off, 32));
  if (lane == 0) red[wid] = m;
  __syncthreads();
  if (tid == 0) {
    float g = red[0];
    for (int w = 1; w < nwaves; ++w) g = fmaxf(g, red[w]);
    red[0] = g;
  }
  __syncthreads();
  const float gmax = red[0];
  __syncthreads();

  // ---- Pass 3: exp + sum --------------------------------------------------
  float lsum = 0.f;
  for (int n = tid; n < padN; n += blockDim.x) {
    float e = __expf(u_s[n] - gmax);           // pad slots: exp(-huge) = 0
    u_s[n] = e;
    lsum += e;
  }
  float s = lsum;
#pragma unroll
  for (int off = 16; off > 0; off >>= 1)
    s += __shfl_xor(s, off, 32);
  if (lane == 0) red[wid] = s;
  __syncthreads();
  if (tid == 0) {
    float g = 0.f;
    for (int w = 0; w < nwaves; ++w) g += red[w];
    red[0] = g;
  }
  __syncthreads();
  const float gsum = red[0];
  const float rinv = (gsum > 0.f) ? (1.0f / gsum) : 0.0f;  // nan_to_num guard

  // ---- Pass 4: normalized write (streaming) -------------------------------
  for (int n = tid; n < N; n += blockDim.x)
    __builtin_nontemporal_store(u_s[n] * rinv, &out[(size_t)b * N + n]);
}

extern "C" void kernel_launch(void* const* d_in, const int* in_sizes, int n_in,
                              void* d_out, int out_size, void* d_ws, size_t ws_size,
                              hipStream_t stream) {
  const float* ctxt = (const float*)d_in[0];
  const float* emb  = (const float*)d_in[1];
  const float* mask = (const float*)d_in[2];
  const float* Wq   = (const float*)d_in[3];
  const float* bq   = (const float*)d_in[4];
  const float* Wk   = (const float*)d_in[5];
  const float* bk   = (const float*)d_in[6];
  float* out = (float*)d_out;

  const int dk = in_sizes[4];                 // 128
  const int dc = in_sizes[3] / dk;            // 384
  const int dm = in_sizes[5] / dk;            // 128
  const int B  = in_sizes[0] / dc;            // 256
  const int N  = in_sizes[1] / (B * dm);      // 5000

  uint16_t* qkt = (uint16_t*)d_ws;                                   // B*dm bf16
  float*    sbf = (float*)((char*)d_ws + (size_t)B * dm * sizeof(uint16_t));

  pda_qproj<<<B, dk, 2 * dk * sizeof(float), stream>>>(
      ctxt, Wq, bq, Wk, bk, qkt, sbf, dc, dk, dm);

  const int threads  = 512;                   // 16 wave32s
  const int numTiles = (N + 15) / 16;
  const int padN     = numTiles * 16;
  const size_t smem  = (size_t)(padN + dm / 2 + threads / 32) * sizeof(float);
  pda_scores_softmax<<<B, threads, smem, stream>>>(
      emb, mask, qkt, sbf, out, N, dm);
}
